// GraphAttentionLayer_472446402908
// MI455X (gfx1250) — compile-verified, matrix operand order checked by reading, not run on hardware
//
#include <hip/hip_runtime.h>
#include <hip/hip_bf16.h>
#include <float.h>

// ---------------------------------------------------------------------------
// Problem constants (reference: B=8, N=1024, FIN=128, E=128, OUT=128, NH=8, HD=16)
// ---------------------------------------------------------------------------
#define BB   8
#define NN   1024
#define FIN  128
#define EE   128
#define OUTD 128
#define NH   8
#define HD   16
#define ROWS (BB * NN)          // 8192

typedef __attribute__((ext_vector_type(16))) _Float16 v16h;
typedef __attribute__((ext_vector_type(8)))  _Float16 v8h;
typedef __attribute__((ext_vector_type(8)))  float    v8f;

static __device__ inline v16h cat16(v8h lo, v8h hi) {
    return __builtin_shufflevector(lo, hi, 0,1,2,3,4,5,6,7,8,9,10,11,12,13,14,15);
}
static __device__ inline v8f wmma_f16(v16h a, v16h b, v8f c) {
    // D = A(16x32 f16) * B(32x16 f16) + C(16x16 f32)
    return __builtin_amdgcn_wmma_f32_16x16x32_f16(false, a, false, b, (short)0, c, false, false);
}

// ---------------------------------------------------------------------------
// Elementwise prep kernels
// ---------------------------------------------------------------------------
__global__ void f32_to_f16_kernel(const float* __restrict__ in, _Float16* __restrict__ out, int n) {
    int i = blockIdx.x * blockDim.x + threadIdx.x;
    if (i < n) out[i] = (_Float16)in[i];
}

// WT[n*K + k] = (f16) W[k*N + n]   (store B-operand as N x K, rows contiguous in K)
__global__ void transpose_to_f16_kernel(const float* __restrict__ W, _Float16* __restrict__ WT,
                                        int K, int N) {
    int i = blockIdx.x * blockDim.x + threadIdx.x;
    if (i < N * K) {
        int n = i / K, k = i % K;
        WT[i] = (_Float16)W[(size_t)k * N + n];
    }
}

// vT[b][h][d][m] = qkv[(b*N+m)*384 + 256 + h*16 + d]
__global__ void transpose_v_kernel(const _Float16* __restrict__ qkv, _Float16* __restrict__ vT) {
    int i = blockIdx.x * blockDim.x + threadIdx.x;
    if (i < BB * NH * HD * NN) {
        int m = i & (NN - 1);
        int d = (i >> 10) & (HD - 1);
        int h = (i >> 14) & (NH - 1);
        int b = i >> 17;
        vT[i] = qkv[((size_t)(b * NN + m)) * (3 * EE) + 2 * EE + h * HD + d];
    }
}

// Fold out_proj + final linear (both affine, no nonlinearity between):
//   out = o @ (W_out @ W_fin) + (b_out @ W_fin + b_fin)
// WcT[n*128 + k] = (f16) sum_j W_out[k][j] * W_fin[j][n] ;  bc[n] = sum_j b_out[j]*W_fin[j][n] + b_fin[n]
__global__ void fold_out_weights_kernel(const float* __restrict__ W_out, const float* __restrict__ b_out,
                                        const float* __restrict__ W_fin, const float* __restrict__ b_fin,
                                        _Float16* __restrict__ WcT, float* __restrict__ bc) {
    int i = blockIdx.x * blockDim.x + threadIdx.x;
    if (i >= EE * OUTD) return;
    int n = i / EE, k = i % EE;
    float s = 0.f;
    for (int j = 0; j < EE; ++j) s += W_out[k * OUTD + j] * W_fin[j * OUTD + n];
    WcT[i] = (_Float16)s;
    if (k == 0) {
        float bs = b_fin[n];
        for (int j = 0; j < EE; ++j) bs += b_out[j] * W_fin[j * OUTD + n];
        bc[n] = bs;
    }
}

// ---------------------------------------------------------------------------
// Generic WMMA GEMM: C[M x N] = A[M x K] * BT[N x K]^T + bias[N]
// One wave per 16x16 output tile, K in steps of 32.
// ---------------------------------------------------------------------------
template <int OUT_F16>
__global__ __launch_bounds__(128) void gemm_bias_kernel(
    const _Float16* __restrict__ A, const _Float16* __restrict__ BT,
    const float* __restrict__ bias, void* __restrict__ C, int M, int N, int K) {
    int tid = threadIdx.x;
    int w   = blockIdx.x * (blockDim.x >> 5) + (tid >> 5);
    int ntn = N >> 4;
    if (w >= (M >> 4) * ntn) return;              // whole wave exits together (EXEC stays full)
    int tm = w / ntn, tn = w % ntn;
    int l = tid & 31, lr = l & 15, lh = l >> 4;

    const _Float16* arow = A  + (size_t)(tm * 16 + lr) * K;
    const _Float16* brow = BT + (size_t)(tn * 16 + lr) * K;
    v8f acc = {};
#pragma unroll 4
    for (int k0 = 0; k0 < K; k0 += 32) {
        v8h alo = *(const v8h*)(arow + k0 + 8 * lh);
        v8h ahi = *(const v8h*)(arow + k0 + 16 + 8 * lh);
        v8h blo = *(const v8h*)(brow + k0 + 8 * lh);
        v8h bhi = *(const v8h*)(brow + k0 + 16 + 8 * lh);
        acc = wmma_f16(cat16(alo, ahi), cat16(blo, bhi), acc);
    }
    int   n  = tn * 16 + lr;
    float bv = bias[n];
    if (OUT_F16) {
        _Float16* out = (_Float16*)C;
#pragma unroll
        for (int v = 0; v < 8; ++v)
            out[(size_t)(tm * 16 + v + 8 * lh) * N + n] = (_Float16)(acc[v] + bv);
    } else {
        float* out = (float*)C;
#pragma unroll
        for (int v = 0; v < 8; ++v)
            out[(size_t)(tm * 16 + v + 8 * lh) * N + n] = acc[v] + bv;
    }
}

// ---------------------------------------------------------------------------
// Kernel: attn_w[b, n0:n0+16, :] = softmax(leakyrelu(Hp Hp^T) * A)   (f16 out)
// grid = B * (N/16) workgroups, 128 threads (4 waves). 16x1024 f16 strip in LDS.
// ---------------------------------------------------------------------------
__global__ __launch_bounds__(128) void adj_attn_kernel(
    const _Float16* __restrict__ Hp, const float* __restrict__ Aadj,
    _Float16* __restrict__ attn_w) {
    __shared__ alignas(16) _Float16 s16[16 * NN];   // 32 KB score strip
    int tid = threadIdx.x;
    int wv = tid >> 5, l = tid & 31, lr = l & 15, lh = l >> 4;
    int wg = blockIdx.x;
    int b = wg >> 6, n0 = (wg & 63) * 16;

    const _Float16* base = Hp + (size_t)b * NN * EE;
    const _Float16* arow = base + (size_t)(n0 + lr) * EE;

    v16h a[4];
#pragma unroll
    for (int kt = 0; kt < 4; ++kt) {
        int k0 = kt * 32;
        v8h lo = *(const v8h*)(arow + k0 + 8 * lh);
        v8h hi = *(const v8h*)(arow + k0 + 16 + 8 * lh);
        a[kt] = cat16(lo, hi);
    }
    for (int ct = 0; ct < 16; ++ct) {
        int m0 = wv * 256 + ct * 16;
        const _Float16* brow = base + (size_t)(m0 + lr) * EE;
        v8f acc = {};
#pragma unroll
        for (int kt = 0; kt < 4; ++kt) {
            int k0 = kt * 32;
            v8h lo = *(const v8h*)(brow + k0 + 8 * lh);
            v8h hi = *(const v8h*)(brow + k0 + 16 + 8 * lh);
            acc = wmma_f16(a[kt], cat16(lo, hi), acc);
        }
#pragma unroll
        for (int v = 0; v < 8; ++v)
            s16[(v + 8 * lh) * NN + m0 + lr] = (_Float16)acc[v];
    }
    __syncthreads();

    // 8 lanes per row (contiguous within a wave), 128 cols each
    int r = tid >> 3, sub = tid & 7, c0 = sub * 128;
    const float* arowA = Aadj + ((size_t)b * NN + n0 + r) * NN;
    float mx = -FLT_MAX;
    for (int c = c0; c < c0 + 128; ++c) {
        float x = (float)s16[r * NN + c];
        x = (x >= 0.f) ? x : 0.2f * x;
        x *= arowA[c];
        s16[r * NN + c] = (_Float16)x;
        mx = fmaxf(mx, x);
    }
    mx = fmaxf(mx, __shfl_xor(mx, 1, 32));
    mx = fmaxf(mx, __shfl_xor(mx, 2, 32));
    mx = fmaxf(mx, __shfl_xor(mx, 4, 32));
    float sum = 0.f;
    for (int c = c0; c < c0 + 128; ++c) {
        float e = __expf((float)s16[r * NN + c] - mx);
        s16[r * NN + c] = (_Float16)e;
        sum += e;
    }
    sum += __shfl_xor(sum, 1, 32);
    sum += __shfl_xor(sum, 2, 32);
    sum += __shfl_xor(sum, 4, 32);
    float inv = 1.0f / sum;
    _Float16* orow = attn_w + ((size_t)b * NN + n0 + r) * NN;
    for (int c = c0; c < c0 + 128; ++c) orow[c] = (_Float16)((float)s16[r * NN + c] * inv);
}

// ---------------------------------------------------------------------------
// Fused MHA kernel: per (b, h, 16-row tile):
//   scores = q k^T / 4 + attn_w ; p = softmax(scores) ; o = p v
// grid = B*(N/16)*NH (head fastest => attn_w strip reused by 8 consecutive blocks).
// 128 threads. 16x1024 f16 score strip in LDS; p A-operand read back as b128.
// ---------------------------------------------------------------------------
__global__ __launch_bounds__(128) void mha_kernel(
    const _Float16* __restrict__ qkv, const _Float16* __restrict__ vT,
    const _Float16* __restrict__ attn_w, _Float16* __restrict__ o) {
    __shared__ alignas(16) _Float16 s16[16 * NN];   // 32 KB score / p strip
    int tid = threadIdx.x;
    int wv = tid >> 5, l = tid & 31, lr = l & 15, lh = l >> 4;
    int wg = blockIdx.x;                 // wg = ((b*64 + nt)*8 + h)
    int h  = wg & 7;
    int nt = (wg >> 3) & 63;
    int b  = wg >> 9;
    int n0 = nt * 16;

    v8h z = {};
    // A tile = q rows (K = HD = 16, zero-padded to 32)
    const _Float16* qrow =
        qkv + ((size_t)(b * NN + n0 + lr)) * (3 * EE) + h * HD + 8 * lh;
    v16h aq = cat16(*(const v8h*)qrow, z);

    for (int ct = 0; ct < 16; ++ct) {
        int m0 = wv * 256 + ct * 16;
        const _Float16* krow =
            qkv + ((size_t)(b * NN + m0 + lr)) * (3 * EE) + EE + h * HD + 8 * lh;
        v8f acc = {};
        acc = wmma_f16(aq, cat16(*(const v8h*)krow, z), acc);
#pragma unroll
        for (int v = 0; v < 8; ++v)
            s16[(v + 8 * lh) * NN + m0 + lr] = (_Float16)acc[v];
    }
    __syncthreads();

    // softmax(scores*0.25 + attn_w) per row; normalized p left in LDS (f16)
    int r = tid >> 3, sub = tid & 7, c0 = sub * 128;
    const _Float16* awrow = attn_w + ((size_t)b * NN + n0 + r) * NN;
    float mx = -FLT_MAX;
    for (int c = c0; c < c0 + 128; ++c) {
        float x = (float)s16[r * NN + c] * 0.25f + (float)awrow[c];
        s16[r * NN + c] = (_Float16)x;
        mx = fmaxf(mx, x);
    }
    mx = fmaxf(mx, __shfl_xor(mx, 1, 32));
    mx = fmaxf(mx, __shfl_xor(mx, 2, 32));
    mx = fmaxf(mx, __shfl_xor(mx, 4, 32));
    float sum = 0.f;
    for (int c = c0; c < c0 + 128; ++c) {
        float e = __expf((float)s16[r * NN + c] - mx);
        s16[r * NN + c] = (_Float16)e;
        sum += e;
    }
    sum += __shfl_xor(sum, 1, 32);
    sum += __shfl_xor(sum, 2, 32);
    sum += __shfl_xor(sum, 4, 32);
    float inv = 1.0f / sum;
    for (int c = c0; c < c0 + 128; ++c)
        s16[r * NN + c] = (_Float16)((float)s16[r * NN + c] * inv);
    __syncthreads();

    // o partial: each wave contracts its 256-col slice (8 k-tiles of 32);
    // p A-operand loads straight from LDS as b128 in WMMA register layout.
    v8f oacc = {};
    const _Float16* vrow = vT + ((size_t)(b * NH + h) * HD + lr) * NN;
    for (int kt = 0; kt < 8; ++kt) {
        int k0 = wv * 256 + kt * 32;
        v8h plo = *(const v8h*)&s16[lr * NN + k0 + 8 * lh];
        v8h phi = *(const v8h*)&s16[lr * NN + k0 + 16 + 8 * lh];
        v8h vlo = *(const v8h*)(vrow + k0 + 8 * lh);
        v8h vhi = *(const v8h*)(vrow + k0 + 16 + 8 * lh);
        oacc = wmma_f16(cat16(plo, phi), cat16(vlo, vhi), oacc);
    }
    __syncthreads();
    // reduce 4 wave partials via LDS (reuse strip as f32 scratch), write o (f16)
    float* po = (float*)s16;  // [wave][16][16] f32
#pragma unroll
    for (int v = 0; v < 8; ++v)
        po[wv * 256 + (v + 8 * lh) * 16 + lr] = oacc[v];
    __syncthreads();
    for (int e = tid; e < 256; e += 128) {
        int row = e >> 4, col = e & 15;
        float s = po[e] + po[256 + e] + po[512 + e] + po[768 + e];
        o[((size_t)(b * NN + n0 + row)) * EE + h * HD + col] = (_Float16)s;
    }
}

// ---------------------------------------------------------------------------
// Host-side launcher
// ---------------------------------------------------------------------------
extern "C" void kernel_launch(void* const* d_in, const int* in_sizes, int n_in,
                              void* d_out, int out_size, void* d_ws, size_t ws_size,
                              hipStream_t stream) {
    const float* H     = (const float*)d_in[0];
    const float* Aadj  = (const float*)d_in[1];
    const float* W_lin = (const float*)d_in[2];
    const float* b_lin = (const float*)d_in[3];
    const float* W_in  = (const float*)d_in[4];
    const float* b_in  = (const float*)d_in[5];
    const float* W_out = (const float*)d_in[6];
    const float* b_out = (const float*)d_in[7];
    const float* W_fin = (const float*)d_in[8];
    const float* b_fin = (const float*)d_in[9];
    float* out = (float*)d_out;

    size_t off = 0;
    auto carve = [&](size_t bytes) {
        void* p = (char*)d_ws + off;
        off += (bytes + 255) & ~(size_t)255;
        return p;
    };
    _Float16* H_h   = (_Float16*)carve((size_t)ROWS * FIN * 2);
    _Float16* WlinT = (_Float16*)carve((size_t)EE * FIN * 2);
    _Float16* WinT  = (_Float16*)carve((size_t)3 * EE * EE * 2);
    _Float16* WcT   = (_Float16*)carve((size_t)OUTD * EE * 2);
    float*    bc    = (float*)carve((size_t)OUTD * 4);
    _Float16* Hp_h  = (_Float16*)carve((size_t)ROWS * EE * 2);
    _Float16* qkv_h = (_Float16*)carve((size_t)ROWS * 3 * EE * 2);
    _Float16* vT_h  = (_Float16*)carve((size_t)BB * NH * HD * NN * 2);
    _Float16* attnw = (_Float16*)carve((size_t)BB * NN * NN * 2);
    _Float16* o_h   = (_Float16*)carve((size_t)ROWS * EE * 2);

    // 1) conversions / weight prep
    {
        int n = ROWS * FIN;
        f32_to_f16_kernel<<<(n + 255) / 256, 256, 0, stream>>>(H, H_h, n);
    }
    transpose_to_f16_kernel<<<(EE * FIN + 255) / 256, 256, 0, stream>>>(W_lin, WlinT, FIN, EE);
    transpose_to_f16_kernel<<<(3 * EE * EE + 255) / 256, 256, 0, stream>>>(W_in, WinT, EE, 3 * EE);
    fold_out_weights_kernel<<<(EE * OUTD + 255) / 256, 256, 0, stream>>>(W_out, b_out, W_fin, b_fin, WcT, bc);

    // 2) Hp = H @ W_lin + b_lin   (f16 out)
    {
        int tiles = (ROWS / 16) * (EE / 16);
        gemm_bias_kernel<1><<<tiles / 4, 128, 0, stream>>>(H_h, WlinT, b_lin, Hp_h, ROWS, EE, FIN);
    }
    // 3) qkv = Hp @ W_in + b_in   (f16 out), then transpose v
    {
        int tiles = (ROWS / 16) * (3 * EE / 16);
        gemm_bias_kernel<1><<<tiles / 4, 128, 0, stream>>>(Hp_h, WinT, b_in, qkv_h, ROWS, 3 * EE, EE);
        int n = BB * NH * HD * NN;
        transpose_v_kernel<<<(n + 255) / 256, 256, 0, stream>>>(qkv_h, vT_h);
    }
    // 4) attn_w = softmax(leakyrelu(Hp Hp^T) * A)   (f16)
    adj_attn_kernel<<<BB * (NN / 16), 128, 0, stream>>>(Hp_h, Aadj, attnw);
    // 5) fused MHA (heads fastest in grid for attn_w reuse)
    mha_kernel<<<BB * (NN / 16) * NH, 128, 0, stream>>>(qkv_h, vT_h, attnw, o_h);
    // 6) out = o @ (W_out W_fin) + (b_out W_fin + b_fin)   (f32 -> d_out)
    {
        int tiles = (ROWS / 16) * (OUTD / 16);
        gemm_bias_kernel<0><<<tiles / 4, 128, 0, stream>>>(o_h, WcT, bc, out, ROWS, OUTD, EE);
    }
}
